// AttentionBlock_17085379904392
// MI455X (gfx1250) — matrix operand order repaired and checked by reading, not compiled, not run on hardware
//
#include <hip/hip_runtime.h>
#include <hip/hip_bf16.h>
#include <cstdint>

#define DEVI __device__ __forceinline__

typedef __attribute__((ext_vector_type(16))) __bf16 v16bf;
typedef __attribute__((ext_vector_type(8)))  __bf16 v8bf;
typedef __attribute__((ext_vector_type(8)))  float  v8f;

constexpr int Bc = 8;          // batch
constexpr int Cc = 512;        // channels
constexpr int HW = 1024;       // H*W = 32*32
constexpr int NH = 8;          // heads
constexpr int HD = 64;         // head dim
constexpr int GRP = 32;        // groupnorm groups
constexpr int CPG = Cc / GRP;  // 16 channels per group
constexpr float EPSc   = 1e-5f;
constexpr float SCALEc = 0.125f; // 64^-0.5

// ---------- helpers ----------

DEVI __bf16 f2bf(float f) {
  unsigned u = __builtin_bit_cast(unsigned, f);
  u += 0x7FFFu + ((u >> 16) & 1u);                 // round-to-nearest-even
  unsigned short s = (unsigned short)(u >> 16);
  return __builtin_bit_cast(__bf16, s);
}

DEVI v16bf mk16(v8bf lo, v8bf hi) {
  v16bf r;
#pragma unroll
  for (int i = 0; i < 8; ++i) { r[i] = lo[i]; r[i + 8] = hi[i]; }
  return r;
}

// A-matrix fragment (16-bit, 16x32, M=lane&15): caller passes row_base + kbase
// (kbase = 0 for lanes 0-15, 16 for lanes 16-31). 16 contiguous K elements.
DEVI v16bf ldA(const __bf16* p) {
  v8bf lo = *reinterpret_cast<const v8bf*>(p);
  v8bf hi = *reinterpret_cast<const v8bf*>(p + 8);
  return mk16(lo, hi);
}

// B-matrix fragment (16-bit, 32x16, N=lane&15): caller passes col_base + koff
// (koff = 0 for lanes 0-15, 16 for lanes 16-31). 16 contiguous K elements.
DEVI v16bf ldB(const __bf16* p) {
  v8bf lo = *reinterpret_cast<const v8bf*>(p);
  v8bf hi = *reinterpret_cast<const v8bf*>(p + 8);
  return mk16(lo, hi);
}

DEVI v8f wmma_bf16(v16bf a, v16bf b, v8f c) {
  return __builtin_amdgcn_wmma_f32_16x16x32_bf16(false, a, false, b,
                                                 (short)0, c, false, false);
}

// max / sum across the 16-lane half (rows m=r / m=r+8 of the C/D layout)
DEVI float halfmax(float v) {
  v = fmaxf(v, __shfl_xor(v, 1, 32));
  v = fmaxf(v, __shfl_xor(v, 2, 32));
  v = fmaxf(v, __shfl_xor(v, 4, 32));
  v = fmaxf(v, __shfl_xor(v, 8, 32));
  return v;
}
DEVI float halfsum(float v) {
  v += __shfl_xor(v, 1, 32);
  v += __shfl_xor(v, 2, 32);
  v += __shfl_xor(v, 4, 32);
  v += __shfl_xor(v, 8, 32);
  return v;
}

// async 16B global -> LDS copy (per lane), tracked by ASYNCcnt
DEVI void async_cp16(unsigned lds_byte_off, const void* gaddr) {
  uint64_t ga = (uint64_t)(uintptr_t)gaddr;
  asm volatile("global_load_async_to_lds_b128 %0, %1, off"
               :: "v"(lds_byte_off), "v"(ga) : "memory");
}
DEVI void wait_async0() {
  asm volatile("s_wait_asynccnt 0" ::: "memory");
}

// ---------- kernel 1: f32 -> bf16 weight conversion ----------

__global__ void to_bf16_kernel(const float* __restrict__ src,
                               __bf16* __restrict__ dst, int n) {
  int i = blockIdx.x * blockDim.x + threadIdx.x;
  if (i < n) dst[i] = f2bf(src[i]);
}

// ---------- kernel 2: GroupNorm, writes hT[b][hw][c] (bf16, c contiguous) ----------

__global__ void __launch_bounds__(256) groupnorm_kernel(
    const float* __restrict__ x, const float* __restrict__ gamma,
    const float* __restrict__ beta, __bf16* __restrict__ hT) {
  int bg = blockIdx.x;
  int b = bg / GRP, g = bg % GRP;
  int t = threadIdx.x;
  const int N = CPG * HW; // 16384 elements per (b, group)
  const float* xb = x + ((size_t)b * Cc + (size_t)g * CPG) * HW;

  float s1 = 0.f, s2 = 0.f;
  for (int i = t; i < N; i += 256) {
    float v = xb[i];
    s1 += v; s2 += v * v;
  }
  __shared__ float r1[256], r2[256];
  r1[t] = s1; r2[t] = s2;
  __syncthreads();
  for (int off = 128; off > 0; off >>= 1) {
    if (t < off) { r1[t] += r1[t + off]; r2[t] += r2[t + off]; }
    __syncthreads();
  }
  float mean = r1[0] * (1.f / N);
  float var  = r2[0] * (1.f / N) - mean * mean;
  float rinv = rsqrtf(var + EPSc);

  for (int i = t; i < N; i += 256) {
    int cl = i >> 10, hw = i & (HW - 1);
    int c = g * CPG + cl;
    float v = (xb[i] - mean) * rinv * gamma[c] + beta[c];
    hT[((size_t)b * HW + hw) * Cc + c] = f2bf(v);
  }
}

// ---------- kernel 3: QKV GEMM (4x4 WMMA register tile per wave) ----------
// out[o][hw] = sum_c w[o][c] * hT[b][hw][c] + bias[o]
// q,k -> [b][h][hw][hd] (q pre-scaled by SCALE), v -> [b][h][hd][hw]

__global__ void __launch_bounds__(32) qkv_gemm_kernel(
    const __bf16* __restrict__ w, const float* __restrict__ bias,
    const __bf16* __restrict__ hT, __bf16* __restrict__ qb,
    __bf16* __restrict__ kbuf, __bf16* __restrict__ vb) {
  int lane = threadIdx.x;
  int lo = lane & 15, hi = lane >> 4;
  int kbA = hi ? 16 : 0, kbB = hi ? 16 : 0;

  int blk = blockIdx.x;
  int hwp = blk % 16; blk /= 16;  // 16 quads of 16-wide hw tiles
  int op  = blk % 24; blk /= 24;  // 24 quads of 16-wide o tiles (3C = 1536)
  int b   = blk;
  int o0 = op * 64, hw0 = hwp * 64;

  v8f acc[4][4] = {};
  for (int c0 = 0; c0 < Cc; c0 += 32) {
    v16bf a[4], bbv[4];
#pragma unroll
    for (int oi = 0; oi < 4; ++oi)
      a[oi] = ldA(w + (size_t)(o0 + oi * 16 + lo) * Cc + c0 + kbA);
#pragma unroll
    for (int hj = 0; hj < 4; ++hj)
      bbv[hj] = ldB(hT + ((size_t)b * HW + hw0 + hj * 16 + lo) * Cc + c0 + kbB);
#pragma unroll
    for (int oi = 0; oi < 4; ++oi)
#pragma unroll
      for (int hj = 0; hj < 4; ++hj)
        acc[oi][hj] = wmma_bf16(a[oi], bbv[hj], acc[oi][hj]);
  }

  // epilogue: segment (q/k/v) and head are uniform per 16-wide o tile
#pragma unroll
  for (int oi = 0; oi < 4; ++oi) {
    int otile = o0 + oi * 16;
    int seg = otile >> 9;            // 0=q, 1=k, 2=v
    int ob = otile & (Cc - 1);
    int h = ob >> 6, dbase = ob & 63;
    __bf16* qk = (seg == 0) ? qb : kbuf;
    float scl = (seg == 0) ? SCALEc : 1.0f;
    if (seg < 2) {
      __bf16* base = qk + (((size_t)b * NH + h) * HW) * HD + dbase;
#pragma unroll
      for (int r = 0; r < 8; ++r) {
        int d = r + 8 * hi;
        float bia = bias[otile + d];
#pragma unroll
        for (int hj = 0; hj < 4; ++hj) {
          int hw = hw0 + hj * 16 + lo;
          base[(size_t)hw * HD + d] = f2bf((acc[oi][hj][r] + bia) * scl);
        }
      }
    } else {
      __bf16* base = vb + (((size_t)b * NH + h) * HD + dbase) * HW;
#pragma unroll
      for (int r = 0; r < 8; ++r) {
        int d = r + 8 * hi;
        float bia = bias[otile + d];
#pragma unroll
        for (int hj = 0; hj < 4; ++hj) {
          int hw = hw0 + hj * 16 + lo;
          base[(size_t)d * HW + hw] = f2bf(acc[oi][hj][r] + bia);
        }
      }
    }
  }
}

// ---------- kernel 4: flash attention ----------
// 4 waves per block, one 16-row q tile each; K/V tiles for 32 keys staged
// cooperatively into LDS with global_load_async_to_lds_b128 (ASYNCcnt).

constexpr int QW = 4; // q tiles (waves) per block

__global__ void __launch_bounds__(32 * QW) attention_kernel(
    const __bf16* __restrict__ qb, const __bf16* __restrict__ kbuf,
    const __bf16* __restrict__ vb, __bf16* __restrict__ aT) {
  __shared__ __align__(16) __bf16 Klds[32 * HD];       // 4 KB, [kj][d]
  __shared__ __align__(16) __bf16 Vlds[HD * 32];       // 4 KB, [d][kj]
  __shared__ __align__(16) __bf16 Plds[QW][16 * 32];   // 4 KB, per-wave P

  int tid = threadIdx.x;
  int lane = tid & 31, wv = tid >> 5;
  int lo = lane & 15, hi = lane >> 4;
  int kbA = hi ? 16 : 0, kbB = hi ? 16 : 0;

  int blk = blockIdx.x;
  int qg = blk % (HW / (16 * QW)); blk /= (HW / (16 * QW));
  int h  = blk % NH;               blk /= NH;
  int b  = blk;
  int q0 = (qg * QW + wv) * 16;

  const __bf16* qbase = qb   + ((size_t)b * NH + h) * HW * HD;
  const __bf16* kbase = kbuf + ((size_t)b * NH + h) * HW * HD;
  const __bf16* vbase = vb   + ((size_t)b * NH + h) * HD * HW;

  v16bf aq0 = ldA(qbase + (size_t)(q0 + lo) * HD + 0  + kbA);
  v16bf aq1 = ldA(qbase + (size_t)(q0 + lo) * HD + 32 + kbA);

  unsigned kOff = (unsigned)(uintptr_t)(&Klds[0]);
  unsigned vOff = (unsigned)(uintptr_t)(&Vlds[0]);

  float mrun[8], lrun[8];
#pragma unroll
  for (int r = 0; r < 8; ++r) { mrun[r] = -3.0e38f; lrun[r] = 0.f; }
  v8f acc[4] = {};

  for (int kb0 = 0; kb0 < HW; kb0 += 32) {
    // ---- stage K (contiguous 4 KB) and V (64 rows of 64 B) into LDS ----
    const char* kg = (const char*)(kbase + (size_t)kb0 * HD);
    const char* vg = (const char*)vbase + (size_t)kb0 * 2;
#pragma unroll
    for (int u = 0; u < 2; ++u) {
      int ci = tid + u * 128;                 // 256 chunks of 16 B
      async_cp16(kOff + ci * 16, kg + (size_t)ci * 16);
    }
#pragma unroll
    for (int u = 0; u < 2; ++u) {
      int ci = tid + u * 128;                 // row = ci>>2 (d), col = ci&3
      int r = ci >> 2, c = ci & 3;
      async_cp16(vOff + ci * 16, vg + (size_t)r * (HW * 2) + c * 16);
    }
    wait_async0();
    __syncthreads();

    // ---- scores for this wave's 16 q rows x 32 keys ----
    v8f s[2];
#pragma unroll
    for (int sub = 0; sub < 2; ++sub) {
      const __bf16* kr = &Klds[(sub * 16 + lo) * HD];
      v16bf bk0 = ldB(kr + 0  + kbB);
      v16bf bk1 = ldB(kr + 32 + kbB);
      v8f sv = {};
      sv = wmma_bf16(aq0, bk0, sv);
      sv = wmma_bf16(aq1, bk1, sv);
      s[sub] = sv;
    }

    // ---- online softmax ----
    float corr[8];
#pragma unroll
    for (int r = 0; r < 8; ++r) {
      float tmax = fmaxf(halfmax(s[0][r]), halfmax(s[1][r]));
      float mnew = fmaxf(mrun[r], tmax);
      float c  = __expf(mrun[r] - mnew);
      float p0 = __expf(s[0][r] - mnew);
      float p1 = __expf(s[1][r] - mnew);
      lrun[r] = lrun[r] * c + halfsum(p0) + halfsum(p1);
      mrun[r] = mnew;
      corr[r] = c;
      int m = r + 8 * hi;
      Plds[wv][m * 32 + lo]      = f2bf(p0);
      Plds[wv][m * 32 + 16 + lo] = f2bf(p1);
    }
#pragma unroll
    for (int t = 0; t < 4; ++t)
#pragma unroll
      for (int r = 0; r < 8; ++r) acc[t][r] *= corr[r];

    asm volatile("s_wait_dscnt 0" ::: "memory");  // P stores visible to own reads

    // ---- P (A layout) x V from LDS, 4 d tiles ----
    v16bf pf = ldA(&Plds[wv][lo * 32 + kbA]);
#pragma unroll
    for (int t = 0; t < 4; ++t) {
      v16bf bv = ldB(&Vlds[(t * 16 + lo) * 32 + kbB]);
      acc[t] = wmma_bf16(pf, bv, acc[t]);
    }
    __syncthreads();   // protect K/V before next stage overwrites
  }

  float rl[8];
#pragma unroll
  for (int r = 0; r < 8; ++r) rl[r] = 1.0f / lrun[r];
#pragma unroll
  for (int t = 0; t < 4; ++t)
#pragma unroll
    for (int r = 0; r < 8; ++r) {
      int m = r + 8 * hi;
      int d = t * 16 + lo;
      aT[((size_t)b * HW + q0 + m) * Cc + h * HD + d] = f2bf(acc[t][r] * rl[r]);
    }
}

// ---------- kernel 5: output projection + bias + residual (4x4 tile) ----------

__global__ void __launch_bounds__(32) proj_gemm_kernel(
    const __bf16* __restrict__ w, const float* __restrict__ bias,
    const __bf16* __restrict__ aT, const float* __restrict__ x,
    float* __restrict__ out) {
  int lane = threadIdx.x;
  int lo = lane & 15, hi = lane >> 4;
  int kbA = hi ? 16 : 0, kbB = hi ? 16 : 0;

  int blk = blockIdx.x;
  int hwp = blk % 16; blk /= 16;  // 16 quads of hw tiles
  int op  = blk % 8;  blk /= 8;   // 8 quads of o tiles (C = 512)
  int b   = blk;
  int o0 = op * 64, hw0 = hwp * 64;

  v8f acc[4][4] = {};
  for (int c0 = 0; c0 < Cc; c0 += 32) {
    v16bf a[4], bbv[4];
#pragma unroll
    for (int oi = 0; oi < 4; ++oi)
      a[oi] = ldA(w + (size_t)(o0 + oi * 16 + lo) * Cc + c0 + kbA);
#pragma unroll
    for (int hj = 0; hj < 4; ++hj)
      bbv[hj] = ldB(aT + ((size_t)b * HW + hw0 + hj * 16 + lo) * Cc + c0 + kbB);
#pragma unroll
    for (int oi = 0; oi < 4; ++oi)
#pragma unroll
      for (int hj = 0; hj < 4; ++hj)
        acc[oi][hj] = wmma_bf16(a[oi], bbv[hj], acc[oi][hj]);
  }

#pragma unroll
  for (int oi = 0; oi < 4; ++oi) {
#pragma unroll
    for (int r = 0; r < 8; ++r) {
      int o = o0 + oi * 16 + r + 8 * hi;
      float bia = bias[o];
#pragma unroll
      for (int hj = 0; hj < 4; ++hj) {
        int hw = hw0 + hj * 16 + lo;
        size_t idx = ((size_t)b * Cc + o) * HW + hw;
        out[idx] = x[idx] + acc[oi][hj][r] + bia;
      }
    }
  }
}

// ---------- launcher ----------

extern "C" void kernel_launch(void* const* d_in, const int* in_sizes, int n_in,
                              void* d_out, int out_size, void* d_ws, size_t ws_size,
                              hipStream_t stream) {
  (void)in_sizes; (void)n_in; (void)out_size; (void)ws_size;
  const float* x     = (const float*)d_in[0];
  const float* gamma = (const float*)d_in[1];
  const float* beta  = (const float*)d_in[2];
  const float* wqkv  = (const float*)d_in[3];
  const float* bqkv  = (const float*)d_in[4];
  const float* wproj = (const float*)d_in[5];
  const float* bproj = (const float*)d_in[6];
  float* out = (float*)d_out;

  char* ws = (char*)d_ws;
  size_t off = 0;
  __bf16* hT     = (__bf16*)(ws + off); off += (size_t)Bc * HW * Cc * 2;      // 8 MiB
  __bf16* qb     = (__bf16*)(ws + off); off += (size_t)Bc * NH * HW * HD * 2; // 8 MiB
  __bf16* kbuf   = (__bf16*)(ws + off); off += (size_t)Bc * NH * HW * HD * 2; // 8 MiB
  __bf16* vb     = (__bf16*)(ws + off); off += (size_t)Bc * NH * HD * HW * 2; // 8 MiB
  __bf16* aT     = (__bf16*)(ws + off); off += (size_t)Bc * HW * Cc * 2;      // 8 MiB
  __bf16* wqkvb  = (__bf16*)(ws + off); off += (size_t)3 * Cc * Cc * 2;       // 1.5 MiB
  __bf16* wprojb = (__bf16*)(ws + off); off += (size_t)Cc * Cc * 2;           // 0.5 MiB

  int nq = 3 * Cc * Cc;
  to_bf16_kernel<<<(nq + 255) / 256, 256, 0, stream>>>(wqkv, wqkvb, nq);
  int np = Cc * Cc;
  to_bf16_kernel<<<(np + 255) / 256, 256, 0, stream>>>(wproj, wprojb, np);

  groupnorm_kernel<<<Bc * GRP, 256, 0, stream>>>(x, gamma, beta, hT);

  qkv_gemm_kernel<<<Bc * 24 * 16, 32, 0, stream>>>(wqkvb, bqkv, hT, qb, kbuf, vb);

  attention_kernel<<<Bc * NH * (HW / (16 * QW)), 32 * QW, 0, stream>>>(qb, kbuf, vb, aT);

  proj_gemm_kernel<<<Bc * 8 * 16, 32, 0, stream>>>(wprojb, bproj, aT, x, out);
}
